// ASAP_58033598104026
// MI455X (gfx1250) — matrix operand order repaired and checked by reading, not compiled, not run on hardware
//
#include <hip/hip_runtime.h>
#include <hip/hip_bf16.h>

typedef __attribute__((ext_vector_type(16))) _Float16 v16h;
typedef __attribute__((ext_vector_type(8)))  _Float16 v8h;
typedef __attribute__((ext_vector_type(8)))  float    v8f;

#define HIDDEN   64
#define EPB      64      // edges per block (4 M-tiles of 16)
#define TPB      128     // 4 wave32s; wave w owns channels [16w,16w+16)
#define KPAD_MAX 96

// ---------------------------------------------------------------------------
// One MLP stage as a WMMA GEMM: [64 x KPAD] (f16, LDS) x [KPAD x 64] (f16, LDS,
// stored transposed as [N][KPAD]) -> relu/affine -> [64 x 64] f16 in LDS.
// B fragments are hoisted into registers and reused across all 4 M-tiles.
// ---------------------------------------------------------------------------
template<int KPAD>
__device__ __forceinline__ void stage_gemm(
    const _Float16* __restrict__ lIn,
    const _Float16* __restrict__ lWt,
    const float* __restrict__ aff,          // [192]: bias[64], scale[64], shift[64]
    _Float16* __restrict__ lOut,            // [64][64]
    int lane, int wave)
{
  constexpr int KSTEPS = KPAD / 32;
  const int nBase = wave * 16;
  const int col   = nBase + (lane & 15);    // output channel (B/C column)
  const int hi    = lane >> 4;              // 0: lanes 0-15, 1: lanes 16-31
  const float bs  = aff[col];
  const float sa  = aff[64 + col];
  const float sc  = aff[128 + col];
  const int  mrow = (lane & 15);            // A row within an M-tile

  // Hoist B fragments: lane holds column `col`, K contiguous (+16 for hi lanes)
  v16h Bf[KSTEPS];
#pragma unroll
  for (int ks = 0; ks < KSTEPS; ++ks)
    Bf[ks] = *(const v16h*)(lWt + col * KPAD + ks * 32 + hi * 16);

#pragma unroll
  for (int mBase = 0; mBase < EPB; mBase += 16) {
    v8f acc = {};
    const int row = mBase + mrow;
#pragma unroll
    for (int ks = 0; ks < KSTEPS; ++ks) {
      // A fragment: halves 0..15 = K[kBase+hi*16..+15] (contiguous, matches B)
      const _Float16* pa = lIn + row * KPAD + ks * 32 + hi * 16;
      v8h a_lo = *(const v8h*)(pa);
      v8h a_hi = *(const v8h*)(pa + 8);
      v16h Af = __builtin_shufflevector(a_lo, a_hi,
                  0,1,2,3,4,5,6,7,8,9,10,11,12,13,14,15);
      acc = __builtin_amdgcn_wmma_f32_16x16x32_f16(
                false, Af, false, Bf[ks], (short)0, acc, false, false);
    }
    // Epilogue: h = scale*relu(dot + bias) + shift  (BN folded, eval mode)
#pragma unroll
    for (int i = 0; i < 8; ++i) {
      float v = acc[i] + bs;
      v = v > 0.f ? v : 0.f;
      v = fmaf(sa, v, sc);
      const int r = mBase + i + 8 * hi;     // C layout: VGPR i -> M = i + 8*hi
      lOut[r * HIDDEN + col] = (_Float16)v;
    }
  }
}

// ---------------------------------------------------------------------------
// Fused EdgeConv layer: gather -> 3x (GEMM+ReLU+BN) -> scatter-max (encoded u32)
// K ordering of stage-0 input: [feat_i(FEATDIM), pos_i(3), pos_j-pos_i(3), pad]
// (weights are row-permuted to match by prep_stage) so the dominant feat gather
// is float4 global loads + 16B-aligned LDS stores.
// ---------------------------------------------------------------------------
template<int FEATDIM, int KPAD0>
__global__ void __launch_bounds__(TPB)
edgeconv_wmma(const float* __restrict__ feat,
              const float* __restrict__ pos,
              const int* __restrict__ ei, int E,
              const _Float16* __restrict__ Wt0, const float* __restrict__ aff0,
              const _Float16* __restrict__ Wt1, const float* __restrict__ aff1,
              const _Float16* __restrict__ Wt2, const float* __restrict__ aff2,
              unsigned int* __restrict__ maxbuf)
{
  __shared__ __align__(32) _Float16 lW0[KPAD0 * HIDDEN];
  __shared__ __align__(32) _Float16 lW1[HIDDEN * HIDDEN];
  __shared__ __align__(32) _Float16 lW2[HIDDEN * HIDDEN];
  __shared__ __align__(32) _Float16 bufA[EPB * KPAD0];
  __shared__ __align__(32) _Float16 bufB[EPB * HIDDEN];
  __shared__ int ldsDst[EPB];

  const int tid  = threadIdx.x;
  const int lane = tid & 31;
  const int wave = tid >> 5;

  // Stage weights into LDS (2 halves per u32)
  {
    const unsigned* g0 = (const unsigned*)Wt0; unsigned* l0 = (unsigned*)lW0;
#pragma unroll 4
    for (int i = tid; i < (HIDDEN * KPAD0) >> 1; i += TPB) l0[i] = g0[i];
    const unsigned* g1 = (const unsigned*)Wt1; unsigned* l1 = (unsigned*)lW1;
    const unsigned* g2 = (const unsigned*)Wt2; unsigned* l2 = (unsigned*)lW2;
#pragma unroll 4
    for (int i = tid; i < (HIDDEN * HIDDEN) >> 1; i += TPB) { l1[i] = g1[i]; l2[i] = g2[i]; }
  }

  // Gather per-edge input rows (f16, K-padded)
  {
    const int e    = tid & (EPB - 1);
    const int part = tid >> 6;              // 2 threads per edge
    const int ge   = blockIdx.x * EPB + e;
    const bool valid = ge < E;
    const int s = valid ? ei[ge]     : 0;   // source j
    const int d = valid ? ei[E + ge] : 0;   // target i (gather + scatter index)
    _Float16* rowp = bufA + e * KPAD0;

    if (FEATDIM == 64) {
      // each part gathers 32 contiguous channels: float4 loads, v8h stores
      const float* fp = feat + d * 64 + part * 32;
#pragma unroll
      for (int i = 0; i < 4; ++i) {
        const float4 f0 = *(const float4*)(fp + 8 * i);
        const float4 f1 = *(const float4*)(fp + 8 * i + 4);
        v8h h;
        h[0] = (_Float16)f0.x; h[1] = (_Float16)f0.y;
        h[2] = (_Float16)f0.z; h[3] = (_Float16)f0.w;
        h[4] = (_Float16)f1.x; h[5] = (_Float16)f1.y;
        h[6] = (_Float16)f1.z; h[7] = (_Float16)f1.w;
        *(v8h*)(rowp + part * 32 + 8 * i) = h;
      }
    } else {
      for (int k = part; k < FEATDIM; k += 2)
        rowp[k] = (_Float16)feat[d * FEATDIM + k];
    }
    if (part == 0) {
      ldsDst[e] = valid ? d : -1;
#pragma unroll
      for (int k = 0; k < 3; ++k) {
        const float pi = pos[d * 3 + k];
        const float pj = pos[s * 3 + k];
        rowp[FEATDIM + k]     = (_Float16)pi;
        rowp[FEATDIM + 3 + k] = (_Float16)(pj - pi);
      }
    }
    for (int k = FEATDIM + 6 + part; k < KPAD0; k += 2)
      rowp[k] = (_Float16)0.f;
  }
  __syncthreads();

  stage_gemm<KPAD0>(bufA, lW0, aff0, bufB, lane, wave);
  __syncthreads();
  stage_gemm<HIDDEN>(bufB, lW1, aff1, bufA, lane, wave);
  __syncthreads();
  stage_gemm<HIDDEN>(bufA, lW2, aff2, bufB, lane, wave);
  __syncthreads();

  // Scatter-max with order-preserving float->u32 encoding (handles negatives)
  for (int i = tid; i < EPB * HIDDEN; i += TPB) {
    const int e  = i >> 6;
    const int ch = i & 63;
    const int d  = ldsDst[e];
    if (d >= 0) {
      const float v = (float)bufB[i];
      unsigned u = __float_as_uint(v);
      u = (u & 0x80000000u) ? ~u : (u | 0x80000000u);
      atomicMax(&maxbuf[d * HIDDEN + ch], u);
    }
  }
}

// ---------------------------------------------------------------------------
// Weight prep: fold BN (eval) into per-channel affine; transpose+pad W to f16.
// Wt layout: [N=64][Kpad]. For stage 0 (fd >= 0) rows are permuted to the
// kernel's K order: knew<fd -> orig 6+knew (feat), else orig knew-fd (pos,pjmi).
// ---------------------------------------------------------------------------
__global__ void prep_stage(const float* __restrict__ W, const float* __restrict__ b,
                           const float* __restrict__ g, const float* __restrict__ be,
                           const float* __restrict__ rm, const float* __restrict__ rv,
                           int K, int Kpad, int fd, _Float16* __restrict__ Wt,
                           float* __restrict__ aff)
{
  const int idx = blockIdx.x * blockDim.x + threadIdx.x;
  const int total = HIDDEN * Kpad;
  if (idx < total) {
    const int n = idx / Kpad;
    const int k = idx % Kpad;
    _Float16 w = (_Float16)0.f;
    if (k < K) {
      int korig = k;
      if (fd >= 0) korig = (k < fd) ? (6 + k) : (k - fd);
      w = (_Float16)W[korig * HIDDEN + n];
    }
    Wt[idx] = w;
  }
  if (idx < HIDDEN) {
    const float a = g[idx] * rsqrtf(rv[idx] + 1e-5f);
    aff[idx]        = b[idx];
    aff[64 + idx]   = a;
    aff[128 + idx]  = be[idx] - rm[idx] * a;
  }
}

__global__ void k_fill_u32(unsigned* __restrict__ p, unsigned v, int n) {
  const int i = blockIdx.x * blockDim.x + threadIdx.x;
  if (i < n) p[i] = v;
}
__global__ void k_zero_f32(float* __restrict__ p, int n) {
  const int i = blockIdx.x * blockDim.x + threadIdx.x;
  if (i < n) p[i] = 0.f;
}
__global__ void k_count(const int* __restrict__ batch, float* __restrict__ cnt, int n) {
  const int i = blockIdx.x * blockDim.x + threadIdx.x;
  if (i < n) atomicAdd(&cnt[batch[i]], 1.f);
}

// Decode max-buffer, ReLU (folds the -inf -> 0 case), write node feats,
// accumulate per-graph mean-pool sums.
__global__ void finalize_kernel(const unsigned* __restrict__ maxbuf,
                                float* __restrict__ feat,
                                float* __restrict__ pooledL,
                                const int* __restrict__ batch, int total)
{
  const int i = blockIdx.x * blockDim.x + threadIdx.x;
  if (i >= total) return;
  const unsigned e = maxbuf[i];
  const unsigned u = (e & 0x80000000u) ? (e & 0x7FFFFFFFu) : ~e;
  float v = __uint_as_float(u);
  v = (v > 0.f && v < 3.0e38f) ? v : 0.f;   // relu + non-finite -> 0
  feat[i] = v;
  atomicAdd(&pooledL[batch[i >> 6] * HIDDEN + (i & 63)], v);
}

// Tiny head: z=[16,256] -> lin1 -> relu -> lin2 -> log_softmax  (~0.3 MFLOP)
__global__ void __launch_bounds__(256)
head_kernel(const float* __restrict__ pooled,   // [4][16][64] sums
            const float* __restrict__ cnt,      // [16]
            const float* __restrict__ w1, const float* __restrict__ b1,
            const float* __restrict__ w2, const float* __restrict__ b2,
            float* __restrict__ out)            // [16][40]
{
  __shared__ float z[16 * 256];
  __shared__ float h1[16 * 64];
  __shared__ float lg[16 * 40];
  const int tid = threadIdx.x;
  for (int i = tid; i < 16 * 256; i += 256) {
    const int gph = i >> 8, k = i & 255, l = k >> 6, c = k & 63;
    float cn = cnt[gph]; cn = cn < 1.f ? 1.f : cn;
    z[i] = pooled[l * (16 * HIDDEN) + gph * HIDDEN + c] / cn;
  }
  __syncthreads();
  for (int i = tid; i < 16 * 64; i += 256) {
    const int gph = i >> 6, j = i & 63;
    float s = b1[j];
    for (int k = 0; k < 256; ++k) s = fmaf(z[gph * 256 + k], w1[k * 64 + j], s);
    h1[i] = s > 0.f ? s : 0.f;
  }
  __syncthreads();
  for (int i = tid; i < 16 * 40; i += 256) {
    const int gph = i / 40, o = i % 40;
    float s = b2[o];
    for (int j = 0; j < 64; ++j) s = fmaf(h1[gph * 64 + j], w2[j * 40 + o], s);
    lg[i] = s;
  }
  __syncthreads();
  if (tid < 16) {
    float m = -3.4e38f;
    for (int o = 0; o < 40; ++o) m = fmaxf(m, lg[tid * 40 + o]);
    float se = 0.f;
    for (int o = 0; o < 40; ++o) se += expf(lg[tid * 40 + o] - m);
    const float ls = logf(se);
    for (int o = 0; o < 40; ++o) out[tid * 40 + o] = lg[tid * 40 + o] - m - ls;
  }
}

// ---------------------------------------------------------------------------
extern "C" void kernel_launch(void* const* d_in, const int* in_sizes, int n_in,
                              void* d_out, int out_size, void* d_ws, size_t ws_size,
                              hipStream_t stream)
{
  (void)n_in; (void)out_size; (void)ws_size;
  const float* x    = (const float*)d_in[0];
  const float* pos  = (const float*)d_in[1];
  const int*   ei   = (const int*)d_in[2];
  const int*   bat  = (const int*)d_in[3];
  const int E = in_sizes[2] / 2;
  const int N = in_sizes[0] / 3;
  const float* lin1W = (const float*)d_in[76];
  const float* lin1b = (const float*)d_in[77];
  const float* lin2W = (const float*)d_in[78];
  const float* lin2b = (const float*)d_in[79];

  // Workspace carve-up (~15.6 MB)
  char* ws = (char*)d_ws;
  size_t off = 0;
  auto alloc = [&](size_t bytes) -> void* {
    void* p = ws + off;
    off = (off + bytes + 255) & ~(size_t)255;
    return p;
  };
  unsigned* maxbuf = (unsigned*)alloc((size_t)N * HIDDEN * 4);
  float* featA  = (float*)alloc((size_t)N * HIDDEN * 4);
  float* featB  = (float*)alloc((size_t)N * HIDDEN * 4);
  float* pooled = (float*)alloc(4 * 16 * HIDDEN * 4);
  float* cnt    = (float*)alloc(16 * 4);
  _Float16* wt[4][3];
  float*    aff[4][3];
  for (int l = 0; l < 4; ++l)
    for (int s = 0; s < 3; ++s) {
      const int Kpad = (s == 0) ? (l == 0 ? 32 : 96) : 64;
      wt[l][s]  = (_Float16*)alloc((size_t)HIDDEN * Kpad * 2);
      aff[l][s] = (float*)alloc(192 * 4);
    }

  // Weight/affine prep for all 12 MLP stages
  for (int l = 0; l < 4; ++l)
    for (int s = 0; s < 3; ++s) {
      const int base = (l == 0) ? (4 + s * 6) : (22 + (l - 1) * 18 + s * 6);
      const int K    = (s == 0) ? (l == 0 ? 9 : 70) : 64;
      const int Kpad = (s == 0) ? (l == 0 ? 32 : 96) : 64;
      const int fd   = (s == 0) ? (K - 6) : -1;   // stage-0 K-permutation
      const int tot  = HIDDEN * Kpad;
      prep_stage<<<(tot + 255) / 256, 256, 0, stream>>>(
          (const float*)d_in[base + 0], (const float*)d_in[base + 1],
          (const float*)d_in[base + 2], (const float*)d_in[base + 3],
          (const float*)d_in[base + 4], (const float*)d_in[base + 5],
          K, Kpad, fd, wt[l][s], aff[l][s]);
    }

  k_zero_f32<<<(4 * 16 * HIDDEN + 255) / 256, 256, 0, stream>>>(pooled, 4 * 16 * HIDDEN);
  k_zero_f32<<<1, 256, 0, stream>>>(cnt, 16);
  k_count<<<(N + 255) / 256, 256, 0, stream>>>(bat, cnt, N);

  const float* fin = x;
  float* fout = featA;
  const int nblk = (E + EPB - 1) / EPB;
  for (int l = 0; l < 4; ++l) {
    k_fill_u32<<<(N * HIDDEN + 255) / 256, 256, 0, stream>>>(
        maxbuf, 0x007FFFFFu /* encode(-inf) */, N * HIDDEN);
    if (l == 0) {
      edgeconv_wmma<3, 32><<<nblk, TPB, 0, stream>>>(
          fin, pos, ei, E,
          wt[l][0], aff[l][0], wt[l][1], aff[l][1], wt[l][2], aff[l][2], maxbuf);
    } else {
      edgeconv_wmma<64, 96><<<nblk, TPB, 0, stream>>>(
          fin, pos, ei, E,
          wt[l][0], aff[l][0], wt[l][1], aff[l][1], wt[l][2], aff[l][2], maxbuf);
    }
    finalize_kernel<<<(N * HIDDEN + 255) / 256, 256, 0, stream>>>(
        maxbuf, fout, pooled + l * 16 * HIDDEN, bat, N * HIDDEN);
    fin  = fout;
    fout = (fout == featA) ? featB : featA;
  }

  head_kernel<<<1, 256, 0, stream>>>(pooled, cnt, lin1W, lin1b, lin2W, lin2b,
                                     (float*)d_out);
}
